// xIDSimilarityMemoryBank_59785944760372
// MI455X (gfx1250) — matrix operand order repaired and checked by reading, not compiled, not run on hardware
//
#include <hip/hip_runtime.h>
#include <math.h>

#define BB 256
#define DD 256
#define MM 131072
#define KK 1024
#define NH 64
#define NS 64
#define NSP 64
#define NSCORE (1 + KK + NS + NSP)   /* 1153 */
#define TEMP_ 0.07f
#define EPSM 1e-4f
#define CHUNK 8192
#define NCHUNK (MM / CHUNK)          /* 16 */

typedef __attribute__((ext_vector_type(16))) _Float16 v16h;
typedef __attribute__((ext_vector_type(8)))  float    v8f;
typedef unsigned int u32x4 __attribute__((ext_vector_type(4)));
typedef int i32x4 __attribute__((ext_vector_type(4)));
typedef int i32x8 __attribute__((ext_vector_type(8)));

#if defined(__has_builtin)
#if __has_builtin(__builtin_amdgcn_tensor_load_to_lds)
#define HAVE_TDM 1
#endif
#endif
#ifndef HAVE_TDM
#define HAVE_TDM 0
#endif

#if HAVE_TDM
// Issue one 2-D TDM tile load (global f32 -> LDS, row-major compacted).
// D# group0: count=1 | lds_addr | global_addr | type=2 ("image")
// D# group1: data_size=4B, tensor dims huge (no OOB), tile dims, row pitch.
__device__ __forceinline__ void tdm_load_tile_2d(unsigned lds_off,
                                                 const float* gptr,
                                                 unsigned tile_cols,
                                                 unsigned tile_rows,
                                                 unsigned row_stride_elems) {
  unsigned long long ga = (unsigned long long)gptr;
  u32x4 g0;
  g0.x = 1u;                                                  // count=1
  g0.y = lds_off;                                             // LDS byte addr
  g0.z = (unsigned)ga;                                        // addr[31:0]
  g0.w = (unsigned)((ga >> 32) & 0x01FFFFFFu) | (2u << 30);   // addr[56:32]|type
  i32x8 g1;
  g1[0] = (int)(2u << 16);                        // data_size code 2 = 4 bytes
  g1[1] = (int)0xFFFF0000u;                       // tensor_dim0[15:0]
  g1[2] = (int)0xFFFFFFFFu;                       // dim0 hi | dim1 lo (huge)
  g1[3] = (int)(0x0000FFFFu | (tile_cols << 16)); // dim1 hi | tile_dim0
  g1[4] = (int)(tile_rows & 0xFFFFu);             // tile_dim1 (tile_dim2=0: 2-D)
  g1[5] = (int)row_stride_elems;                  // tensor_dim0_stride lo
  g1[6] = 0;
  g1[7] = 0;
  i32x4 g2 = 0, g3 = 0;
#if __clang_major__ >= 23
  i32x8 g4 = 0;
  __builtin_amdgcn_tensor_load_to_lds(g0, g1, g2, g3, g4, 0);
#else
  __builtin_amdgcn_tensor_load_to_lds(g0, g1, g2, g3, 0);
#endif
}
#endif

// ---------------------------------------------------------------- normalize
__global__ void k_normalize(const float* __restrict__ video,
                            const float* __restrict__ audio,
                            float* __restrict__ qv, float* __restrict__ qa) {
  __shared__ float red[256];
  int b = blockIdx.x & (BB - 1);
  int p = blockIdx.x >> 8;
  const float* src = p ? audio : video;
  float* dst = p ? qa : qv;
  int e = threadIdx.x;
  float x = src[b * DD + e];
  red[e] = x * x;
  __syncthreads();
  for (int s = 128; s > 0; s >>= 1) {
    if (e < s) red[e] += red[e + s];
    __syncthreads();
  }
  float denom = fmaxf(sqrtf(red[0]), 1e-12f);
  dst[b * DD + e] = x / denom;
}

// ---------------------------------------------------------------- row norms
__global__ void k_rownorm(const float* __restrict__ mem1,
                          const float* __restrict__ mem2,
                          float* __restrict__ rn1, float* __restrict__ rn2) {
  int w = threadIdx.x >> 5, lane = threadIdx.x & 31;
  long long gr = (long long)blockIdx.x * 8 + w;
  int p = gr >= MM;
  long long row = p ? gr - MM : gr;
  const float* mem = p ? mem2 : mem1;
  float* rn = p ? rn2 : rn1;
  const float* src = mem + row * DD + lane * 8;
  float s = 0.f;
#pragma unroll
  for (int j = 0; j < 8; ++j) { float x = src[j]; s += x * x; }
  for (int o = 16; o; o >>= 1) s += __shfl_xor(s, o, 32);
  if (lane == 0) rn[row] = s;
}

// --------------------------------------------------- mining GEMM (f16 WMMA)
// z=0: qv x mem2^T -> scr2 ; z=1: qa x mem1^T -> scr1.
// Block 256 thr (8 waves); tile 16 q-rows x 256 bank rows; wave owns two
// 16x16 N-tiles. TDM double-buffered staging of f32 tiles into LDS.
// Epilogue stores key = dot - 0.5*||m||^2 (L2 top-k ranking key).
__global__ void k_mine_gemm(const float* __restrict__ qv, const float* __restrict__ qa,
                            const float* __restrict__ mem1, const float* __restrict__ mem2,
                            const float* __restrict__ rn1, const float* __restrict__ rn2,
                            float* __restrict__ scr1, float* __restrict__ scr2) {
  __shared__ float Asf[2][16 * 32];
  __shared__ float Bsf[2][256 * 32];
  int z = blockIdx.z;
  const float* q   = z ? qa : qv;
  const float* mem = z ? mem1 : mem2;
  const float* rn  = z ? rn1 : rn2;
  float* scr       = z ? scr1 : scr2;

  int qBase = blockIdx.y * 16;
  long long mBase = (long long)blockIdx.x * 256;
  int t = threadIdx.x;
  int w = t >> 5, lane = t & 31;
  int h = lane >> 4, lrow = lane & 15;

  v8f c0 = 0.0f, c1 = 0.0f;

  auto stage = [&](int buf, int kc) {
#if HAVE_TDM
    // each wave DMAs its own 32-row B subtile; wave 0 also DMAs the A tile.
    // (generic->LDS address: low 32 bits are the workgroup LDS byte offset)
    tdm_load_tile_2d((unsigned)(unsigned long long)(void*)&Bsf[buf][w * 32 * 32],
                     mem + (mBase + w * 32) * DD + kc, 32u, 32u, DD);
    if (w == 0)
      tdm_load_tile_2d((unsigned)(unsigned long long)(void*)&Asf[buf][0],
                       q + (long long)qBase * DD + kc, 32u, 16u, DD);
#else
    const float* src = mem + (mBase + t) * DD + kc;
#pragma unroll
    for (int j = 0; j < 32; ++j) Bsf[buf][t * 32 + j] = src[j];
    int idx = t * 2;
    int rr = idx >> 5, cc2 = idx & 31;
    Asf[buf][idx]     = q[(qBase + rr) * DD + kc + cc2];
    Asf[buf][idx + 1] = q[(qBase + rr) * DD + kc + cc2 + 1];
#endif
  };

  stage(0, 0);
#pragma unroll
  for (int i = 0; i < 8; ++i) {
#if HAVE_TDM
    __builtin_amdgcn_s_wait_tensorcnt(0);
#endif
    __syncthreads();            // buf(i) ready for all; prior reads of buf(i+1) done
    if (i < 7) stage((i + 1) & 1, (i + 1) * 32);
    int buf = i & 1;
    v16h a, b0, b1;
    // A 16x32 f16: lanes 0-15 M=lane, K {0..7,16..23}; lanes 16-31 {8..15,24..31}
#pragma unroll
    for (int e = 0; e < 16; ++e) {
      int ke = e + 8 * h + (e >= 8 ? 8 : 0);
      a[e] = (_Float16)Asf[buf][lrow * 32 + ke];
    }
    // B 32x16 f16: lanes 0-15 K {0..7,16..23}, lanes 16-31 {8..15,24..31}, N=lane%16
#pragma unroll
    for (int e = 0; e < 16; ++e) {
      int kb = e + 8 * h + (e >= 8 ? 8 : 0);
      b0[e] = (_Float16)Bsf[buf][(w * 32 + lrow) * 32 + kb];
      b1[e] = (_Float16)Bsf[buf][(w * 32 + 16 + lrow) * 32 + kb];
    }
    c0 = __builtin_amdgcn_wmma_f32_16x16x32_f16(false, a, false, b0, (short)0, c0, false, false);
    c1 = __builtin_amdgcn_wmma_f32_16x16x32_f16(false, a, false, b1, (short)0, c1, false, false);
  }
  // C layout: VGPR r -> (M = r + 8*h, N = lane%16)
  long long col0 = mBase + w * 32 + lrow;
  long long col1 = col0 + 16;
  float adj0 = 0.5f * rn[col0], adj1 = 0.5f * rn[col1];
#pragma unroll
  for (int r = 0; r < 8; ++r) {
    int row = qBase + r + 8 * h;
    scr[(long long)row * MM + col0] = c0[r] - adj0;
    scr[(long long)row * MM + col1] = c1[r] - adj1;
  }
}

// --------------------------------------------- top-64 phase A (per chunk)
// Load 8192-key chunk into LDS once; 64 argmax rounds entirely in LDS.
__global__ void k_topk_local(const float* __restrict__ scr1, const float* __restrict__ scr2,
                             const int* __restrict__ y,
                             float* __restrict__ candK, int* __restrict__ candI) {
  __shared__ float keys[CHUNK];
  __shared__ float sk[256];
  __shared__ int   si[256];
  int chunk = blockIdx.x, b = blockIdx.y, p = blockIdx.z;  // p=0: scr2->a ; 1: scr1->v
  const float* scr = (p ? scr1 : scr2) + (long long)b * MM + (long long)chunk * CHUNK;
  int ylocal = y[b] - chunk * CHUNK;
  int t = threadIdx.x;
  for (int i = t; i < CHUNK; i += 256)
    keys[i] = (i == ylocal) ? -3.0e38f : scr[i];
  __syncthreads();
  for (int r = 0; r < NH; ++r) {
    float best = -3.0e38f; int bidx = 1 << 30;
    for (int i = t; i < CHUNK; i += 256) {
      float kk = keys[i];
      if (kk > best || (kk == best && i < bidx)) { best = kk; bidx = i; }
    }
    sk[t] = best; si[t] = bidx;
    __syncthreads();
    for (int s = 128; s > 0; s >>= 1) {
      if (t < s) {
        float ok = sk[t + s]; int oi = si[t + s];
        if (ok > sk[t] || (ok == sk[t] && oi < si[t])) { sk[t] = ok; si[t] = oi; }
      }
      __syncthreads();
    }
    if (t == 0) {
      int slot = ((p * BB + b) * NCHUNK + chunk) * NH + r;
      candK[slot] = sk[0];
      candI[slot] = chunk * CHUNK + si[0];
      keys[si[0]] = -3.0e38f;
    }
    __syncthreads();
  }
}

// --------------------------------------------- top-64 phase B (merge)
__global__ void k_topk_merge(const float* __restrict__ candK, const int* __restrict__ candI,
                             int* __restrict__ vhard, int* __restrict__ ahard) {
  __shared__ float keys[NCHUNK * NH];   // 1024
  __shared__ int   gidx[NCHUNK * NH];
  __shared__ float sk[256];
  __shared__ int   si[256];
  __shared__ int   ss[256];
  int b = blockIdx.x, p = blockIdx.y;
  int base = (p * BB + b) * (NCHUNK * NH);
  int t = threadIdx.x;
  for (int i = t; i < NCHUNK * NH; i += 256) { keys[i] = candK[base + i]; gidx[i] = candI[base + i]; }
  __syncthreads();
  int* outInd = (p ? vhard : ahard) + b * NH;
  for (int r = 0; r < NH; ++r) {
    float best = -3.0e38f; int bidx = 1 << 30, bslot = 0;
    for (int i = t; i < NCHUNK * NH; i += 256) {
      float kk = keys[i]; int gi = gidx[i];
      if (kk > best || (kk == best && gi < bidx)) { best = kk; bidx = gi; bslot = i; }
    }
    sk[t] = best; si[t] = bidx; ss[t] = bslot;
    __syncthreads();
    for (int s = 128; s > 0; s >>= 1) {
      if (t < s) {
        float ok = sk[t + s]; int oi = si[t + s];
        if (ok > sk[t] || (ok == sk[t] && oi < si[t])) { sk[t] = ok; si[t] = oi; ss[t] = ss[t + s]; }
      }
      __syncthreads();
    }
    if (t == 0) { outInd[r] = si[0]; keys[ss[0]] = -3.0e38f; }
    __syncthreads();
  }
}

// -------------------------------------------------------------- scores
__global__ void k_scores(const float* __restrict__ qv, const float* __restrict__ qa,
                         const float* __restrict__ mem1, const float* __restrict__ mem2,
                         const int* __restrict__ y, const int* __restrict__ negraw,
                         const int* __restrict__ idx_s, const int* __restrict__ idx_sp,
                         const float* __restrict__ u_s, const float* __restrict__ u_sp,
                         const int* __restrict__ vhard, const int* __restrict__ ahard,
                         float* __restrict__ outScores) {
  int w = threadIdx.x >> 5, lane = threadIdx.x & 31;
  long long gid = (long long)blockIdx.x * 8 + w;
  const long long total = 2LL * BB * NSCORE;
  if (gid >= total) return;
  int p   = (int)(gid / ((long long)BB * NSCORE));
  int rem = (int)(gid % ((long long)BB * NSCORE));
  int b = rem / NSCORE;
  int k = rem % NSCORE;
  const float* q    = (p == 0) ? qv : qa;
  const float* bank = (p == 0) ? mem2 : mem1;
  const int* hard   = (p == 0) ? ahard : vhard;
  int slot          = (p == 0) ? 1 : 0;
  const float* qrow = q + b * DD;
  float score;
  if (k <= KK) {
    int row;
    if (k == 0) row = y[b];
    else { int raw = negraw[b * KK + (k - 1)]; row = raw + (raw >= y[b] ? 1 : 0); }
    const float* mrow = bank + (long long)row * DD;
    float dot = 0.f;
#pragma unroll
    for (int j = 0; j < 8; ++j) { int d = lane + j * 32; dot += mrow[d] * qrow[d]; }
    for (int o = 16; o; o >>= 1) dot += __shfl_xor(dot, o, 32);
    score = dot / TEMP_;
  } else if (k < 1 + KK + NS) {
    int s = k - (1 + KK);
    float u  = u_s[(slot * BB + b) * NS + s];
    float cc = EPSM + u * (1.0f - 2.0f * EPSM);
    int i0 = idx_s[(slot * 2 + 0) * NS + s];
    int i1 = idx_s[(slot * 2 + 1) * NS + s];
    const float* h0 = bank + (long long)hard[b * NH + i0] * DD;
    const float* h1 = bank + (long long)hard[b * NH + i1] * DD;
    float dot = 0.f, nrm = 0.f;
#pragma unroll
    for (int j = 0; j < 8; ++j) {
      int d = lane + j * 32;
      float m = cc * h0[d] + (1.0f - cc) * h1[d];
      dot += m * qrow[d]; nrm += m * m;
    }
    for (int o = 16; o; o >>= 1) { dot += __shfl_xor(dot, o, 32); nrm += __shfl_xor(nrm, o, 32); }
    score = (dot / fmaxf(sqrtf(nrm), 1e-12f)) / TEMP_;
  } else {
    int s = k - (1 + KK + NS);
    float u  = u_sp[(slot * BB + b) * NSP + s];
    float cc = EPSM + u * (0.5f - 2.0f * EPSM);
    const float* h0 = bank + (long long)hard[b * NH + idx_sp[slot * NSP + s]] * DD;
    float dot = 0.f, nrm = 0.f;
#pragma unroll
    for (int j = 0; j < 8; ++j) {
      int d = lane + j * 32;
      float m = cc * qrow[d] + (1.0f - cc) * h0[d];   // emb == query both directions
      dot += m * qrow[d]; nrm += m * m;
    }
    for (int o = 16; o; o >>= 1) { dot += __shfl_xor(dot, o, 32); nrm += __shfl_xor(nrm, o, 32); }
    score = (dot / fmaxf(sqrtf(nrm), 1e-12f)) / TEMP_;
  }
  if (lane == 0) outScores[(long long)p * BB * NSCORE + (long long)b * NSCORE + k] = score;
}

// ----------------------------------------------------- momentum row update
__global__ void k_update(const float* __restrict__ mem1, const float* __restrict__ mem2,
                         const float* __restrict__ qv, const float* __restrict__ qa,
                         const int* __restrict__ y,
                         float* __restrict__ new1, float* __restrict__ new2) {
  __shared__ float red[256];
  int b = blockIdx.x;
  int p = blockIdx.y;
  int yb = y[b];
  for (int b2 = b + 1; b2 < BB; ++b2)   // JAX scatter: last duplicate wins
    if (y[b2] == yb) return;            // uniform per block -> barrier-safe
  const float* mem = p ? mem2 : mem1;
  const float* q   = p ? qa : qv;
  float* dst       = p ? new2 : new1;
  int e = threadIdx.x;
  float t = 0.5f * mem[(long long)yb * DD + e] + 0.5f * q[b * DD + e];
  red[e] = t * t;
  __syncthreads();
  for (int s = 128; s > 0; s >>= 1) {
    if (e < s) red[e] += red[e + s];
    __syncthreads();
  }
  float denom = fmaxf(sqrtf(red[0]), 1e-12f);
  dst[(long long)yb * DD + e] = t / denom;
}

extern "C" void kernel_launch(void* const* d_in, const int* in_sizes, int n_in,
                              void* d_out, int out_size, void* d_ws, size_t ws_size,
                              hipStream_t stream) {
  (void)in_sizes; (void)n_in; (void)out_size; (void)ws_size;
  const float* video  = (const float*)d_in[0];
  const float* audio  = (const float*)d_in[1];
  const float* mem1   = (const float*)d_in[2];
  const float* mem2   = (const float*)d_in[3];
  const int*   y      = (const int*)d_in[4];
  const int*   negraw = (const int*)d_in[5];
  const int*   idx_s  = (const int*)d_in[6];
  const int*   idx_sp = (const int*)d_in[7];
  const float* u_s    = (const float*)d_in[8];
  const float* u_sp   = (const float*)d_in[9];
  // d_in[10] = epoch (20 >= 10: hard-negative path always taken)

  float* out = (float*)d_out;
  float* outScores = out;                               // [2, B, 1153]
  float* new1 = out + 2LL * BB * NSCORE;                // [M, D]
  float* new2 = new1 + (long long)MM * DD;              // [M, D]

  float* ws  = (float*)d_ws;                            // ~5.9 MB total
  float* qv  = ws;
  float* qa  = qv + BB * DD;
  float* rn1 = qa + BB * DD;
  float* rn2 = rn1 + MM;
  int* ahard = (int*)(rn2 + MM);
  int* vhard = ahard + BB * NH;
  float* candK = (float*)(vhard + BB * NH);             // [2,B,16,64]
  int*   candI = (int*)(candK + 2 * BB * NCHUNK * NH);

  // [B,M] == [M,D] in size: reuse output bank regions as GEMM scratch; fully
  // overwritten by the bank copies afterwards.
  float* scr1 = new1;   // qa x mem1 keys
  float* scr2 = new2;   // qv x mem2 keys

  k_normalize<<<2 * BB, 256, 0, stream>>>(video, audio, qv, qa);
  k_rownorm<<<2 * MM / 8, 256, 0, stream>>>(mem1, mem2, rn1, rn2);
  dim3 gg(MM / 256, BB / 16, 2);
  k_mine_gemm<<<gg, 256, 0, stream>>>(qv, qa, mem1, mem2, rn1, rn2, scr1, scr2);
  k_topk_local<<<dim3(NCHUNK, BB, 2), 256, 0, stream>>>(scr1, scr2, y, candK, candI);
  k_topk_merge<<<dim3(BB, 2), 256, 0, stream>>>(candK, candI, vhard, ahard);
  long long total = 2LL * BB * NSCORE;
  int nblk = (int)((total + 7) / 8);
  k_scores<<<nblk, 256, 0, stream>>>(qv, qa, mem1, mem2, y, negraw, idx_s, idx_sp,
                                     u_s, u_sp, vhard, ahard, outScores);
  hipMemcpyAsync(new1, mem1, (size_t)MM * DD * sizeof(float), hipMemcpyDeviceToDevice, stream);
  hipMemcpyAsync(new2, mem2, (size_t)MM * DD * sizeof(float), hipMemcpyDeviceToDevice, stream);
  k_update<<<dim3(BB, 2), 256, 0, stream>>>(mem1, mem2, qv, qa, y, new1, new2);
}
